// Attention_7610682048802
// MI455X (gfx1250) — matrix operand order repaired and checked by reading, not compiled, not run on hardware
//
#include <hip/hip_runtime.h>
#include <hip/hip_bf16.h>

// ---- problem constants -----------------------------------------------------
#define BATCH   256
#define NTOK    196
#define CDIM    384
#define KEYD    32
#define NHEAD   6
#define DHEAD   128                 // value dim per head (ATTN_RATIO*KEY_DIM)
#define DHTOT   768                 // DHEAD*NHEAD
#define HTOT    1152                // DHTOT + 2*KEYD*NHEAD
#define ROWS    (BATCH*NTOK)        // 50176 = 3136*16 (exact M tiling)
#define MTILES  (ROWS/16)           // 3136
#define NPAD    208                 // 196 padded to 13*16 (token dim)
#define KPAD    224                 // 196 padded to 7*32 (WMMA K dim)
#define SPITCH  224
#define QSCALE  0.17677669529663687f  // KEY_DIM^-0.5

typedef __attribute__((ext_vector_type(16))) __bf16 v16bf;
typedef __attribute__((ext_vector_type(8)))  __bf16 bf16x8;
typedef __attribute__((ext_vector_type(8)))  float  v8f;

// ---- WMMA fragment loaders (layouts per CDNA5 ISA 7.12.2, wave32) ----------
// A (16x32 bf16): lane g=(lane>>4): elems 0..7  -> K = g*8 .. g*8+7
//                                   elems 8..15 -> K = 16+g*8 .. 16+g*8+7
// B (32x16 bf16): lane g:           same K mapping as A
__device__ __forceinline__ v16bf load_a_bf16(const __bf16* rowk, int g) {
  bf16x8 lo = *(const bf16x8*)(rowk + g * 8);
  bf16x8 hi = *(const bf16x8*)(rowk + g * 8 + 16);
  return __builtin_shufflevector(lo, hi, 0,1,2,3,4,5,6,7,8,9,10,11,12,13,14,15);
}
__device__ __forceinline__ v16bf load_b_bf16(const __bf16* rowk, int g) {
  bf16x8 lo = *(const bf16x8*)(rowk + g * 8);
  bf16x8 hi = *(const bf16x8*)(rowk + g * 8 + 16);
  return __builtin_shufflevector(lo, hi, 0,1,2,3,4,5,6,7,8,9,10,11,12,13,14,15);
}
__device__ __forceinline__ v16bf load_a_f32(const float* rowk, int g) {
  v8f lo = *(const v8f*)(rowk + g * 8);
  v8f hi = *(const v8f*)(rowk + g * 8 + 16);
  v16bf r;
#pragma unroll
  for (int i = 0; i < 8; ++i) { r[i] = (__bf16)lo[i]; r[i + 8] = (__bf16)hi[i]; }
  return r;
}
#define WMMA_BF16(A, B, C) \
  __builtin_amdgcn_wmma_f32_16x16x32_bf16(false, (A), false, (B), (short)0, (C), false, false)

// ---- kernel 0: one-shot fp32 -> bf16 weight conversion ---------------------
__global__ __launch_bounds__(256) void cvt_kernel(const float* __restrict__ src,
                                                  __bf16* __restrict__ dst, int n) {
  for (int i = blockIdx.x * 256 + threadIdx.x; i < n; i += gridDim.x * 256)
    dst[i] = (__bf16)src[i];
}

// ---- kernel 1: fused QKV GEMM + bias + q-scale + bf16 scatter --------------
// Each wave: 16x64 tile (one A fragment reused by 4 WMMAs per k-step).
__global__ __launch_bounds__(256) void qkv_kernel(
    const float* __restrict__ x, const __bf16* __restrict__ qkv_wb,
    const float* __restrict__ qkv_b,
    __bf16* __restrict__ Q, __bf16* __restrict__ Kb, __bf16* __restrict__ Vt) {
  const int lane = threadIdx.x & 31;
  const int wave = threadIdx.x >> 5;
  const int unit = blockIdx.x * 8 + wave;        // 3136 * 18 units exactly
  const int mt = unit / 18;
  const int nt0 = (unit % 18) * 4;               // first of 4 N-tiles
  const int g = lane >> 4;
  const int r16 = lane & 15;
  const float* arow = x + (size_t)(mt * 16 + r16) * CDIM;

  v8f acc[4] = {};
  for (int kk = 0; kk < CDIM; kk += 32) {
    const v16bf a = load_a_f32(arow + kk, g);
#pragma unroll
    for (int j = 0; j < 4; ++j) {
      const int h = (nt0 + j) * 16 + r16;
      const v16bf b = load_b_bf16(qkv_wb + (size_t)h * CDIM + kk, g);
      acc[j] = WMMA_BF16(a, b, acc[j]);
    }
  }

#pragma unroll
  for (int j = 0; j < 4; ++j) {
    const int h = (nt0 + j) * 16 + r16;          // output column (0..1151)
    const float bias = qkv_b[h];
    const int head = h / 192;
    const int r = h - head * 192;
#pragma unroll
    for (int v = 0; v < 8; ++v) {
      const int row = mt * 16 + v + g * 8;       // C layout: M = v + g*8
      const int bimg = row / NTOK;
      const int tok = row - bimg * NTOK;
      const size_t bh = (size_t)bimg * NHEAD + head;
      const float val = acc[j][v] + bias;
      if (r < KEYD) {
        Q[(bh * NPAD + tok) * KEYD + r] = (__bf16)(val * QSCALE);
      } else if (r < 2 * KEYD) {
        Kb[(bh * NPAD + tok) * KEYD + (r - KEYD)] = (__bf16)val;
      } else {
        Vt[(bh * DHEAD + (r - 2 * KEYD)) * KPAD + tok] = (__bf16)val;
      }
    }
  }
}

// ---- kernel 2: per-(batch,head) attention, 306KB LDS -----------------------
#define S_BYTES  (NPAD * SPITCH * 4)             // 186368: f32 scores
#define P_BYTES  (NPAD * SPITCH * 2)             //  93184: bf16 probs
#define QK_BYTES (NPAD * KEYD * 2)               //  13312: staged Q or K tile
#define SMEM_TOTAL (S_BYTES + P_BYTES + 2 * QK_BYTES)   // 306176 < 320KB WGP LDS
__global__ __launch_bounds__(256) void attn_kernel(
    const __bf16* __restrict__ Q, const __bf16* __restrict__ Kb,
    const __bf16* __restrict__ Vt, const float* __restrict__ attn_biases,
    const int* __restrict__ bias_idxs, __bf16* __restrict__ OutAct) {
  extern __shared__ __attribute__((aligned(64))) char smem_raw[];
  float*  S  = (float*)smem_raw;                        // [208][224]
  __bf16* P  = (__bf16*)(smem_raw + S_BYTES);           // [208][224]
  __bf16* Qs = (__bf16*)(smem_raw + S_BYTES + P_BYTES); // [208][32]
  __bf16* Ks = (__bf16*)(smem_raw + S_BYTES + P_BYTES + QK_BYTES);

  const int lane = threadIdx.x & 31;
  const int wave = threadIdx.x >> 5;
  const int g = lane >> 4;
  const int r16 = lane & 15;
  const int bh = blockIdx.x;
  const int head = bh % NHEAD;
  const int bimg = bh / NHEAD;
  const __bf16* Vp = Vt + (size_t)bh * DHEAD * KPAD;
  const float* biasrow = attn_biases + head * NTOK;

  // Stage Q and K tiles (each re-read 13x below) into LDS: 832 x 16B each.
  {
    const uint4* qg = (const uint4*)(Q + (size_t)bh * NPAD * KEYD);
    const uint4* kg = (const uint4*)(Kb + (size_t)bh * NPAD * KEYD);
    uint4* ql = (uint4*)Qs;
    uint4* kl = (uint4*)Ks;
    for (int i = threadIdx.x; i < QK_BYTES / 16; i += 256) {
      ql[i] = qg[i];
      kl[i] = kg[i];
    }
  }
  __syncthreads();

  // Phase A: S = (Q*scale) K^T + bias.  One A fragment per row, 13 WMMAs.
  for (int tn = wave; tn < 13; tn += 8) {
    const v16bf a = load_a_bf16(Qs + (tn * 16 + r16) * KEYD, g);
    for (int tm = 0; tm < 13; ++tm) {
      const v16bf b = load_b_bf16(Ks + (tm * 16 + r16) * KEYD, g);
      v8f acc = {};
      acc = WMMA_BF16(a, b, acc);
#pragma unroll
      for (int v = 0; v < 8; ++v) {
        const int n = tn * 16 + v + g * 8;
        const int m = tm * 16 + r16;
        float val = acc[v];
        if (n < NTOK && m < NTOK)
          val += biasrow[bias_idxs[n * NTOK + m]];
        S[n * SPITCH + m] = val;
      }
    }
  }
  __syncthreads();

  // Phase B: row softmax, repack to zero-padded bf16 P.
  for (int row = threadIdx.x; row < NPAD; row += 256) {
    if (row < NTOK) {
      float mx = -1e30f;
      for (int m = 0; m < NTOK; ++m) mx = fmaxf(mx, S[row * SPITCH + m]);
      float sum = 0.0f;
      for (int m = 0; m < NTOK; ++m) {
        float e = __expf(S[row * SPITCH + m] - mx);
        S[row * SPITCH + m] = e;
        sum += e;
      }
      const float inv = 1.0f / sum;
      for (int m = 0; m < NTOK; ++m) P[row * SPITCH + m] = (__bf16)(S[row * SPITCH + m] * inv);
      for (int m = NTOK; m < KPAD; ++m) P[row * SPITCH + m] = (__bf16)0.0f;
    } else {
      for (int m = 0; m < KPAD; ++m) P[row * SPITCH + m] = (__bf16)0.0f;  // pad rows
    }
  }
  __syncthreads();

  // Phase C: O = P @ V.  One P fragment reused across all 8 head-dim tiles.
  for (int tn = wave; tn < 13; tn += 8) {
    v8f acc[8] = {};
    for (int kk = 0; kk < KPAD; kk += 32) {
      const v16bf a = load_a_bf16(P + (tn * 16 + r16) * SPITCH + kk, g);  // LDS
#pragma unroll
      for (int td = 0; td < 8; ++td) {
        const v16bf b = load_b_bf16(Vp + (td * 16 + r16) * KPAD + kk, g);
        acc[td] = WMMA_BF16(a, b, acc[td]);
      }
    }
#pragma unroll
    for (int td = 0; td < 8; ++td) {
#pragma unroll
      for (int v = 0; v < 8; ++v) {
        const int tok = tn * 16 + v + g * 8;
        if (tok < NTOK) {
          const int d = td * 16 + r16;
          const float o = acc[td][v];
          const float hs = o * fminf(fmaxf(o + 3.0f, 0.0f), 6.0f) * (1.0f / 6.0f);
          OutAct[((size_t)bimg * NTOK + tok) * DHTOT + head * DHEAD + d] = (__bf16)hs;
        }
      }
    }
  }
}

// ---- kernel 3: projection GEMM, fp32 output --------------------------------
// Each wave: 16x64 tile (one A fragment reused by 4 WMMAs per k-step).
__global__ __launch_bounds__(256) void proj_kernel(
    const __bf16* __restrict__ OutAct, const __bf16* __restrict__ proj_wb,
    const float* __restrict__ proj_b, float* __restrict__ out) {
  const int lane = threadIdx.x & 31;
  const int wave = threadIdx.x >> 5;
  const int unit = blockIdx.x * 8 + wave;        // 3136 * 6 units exactly
  const int mt = unit / 6;
  const int nt0 = (unit % 6) * 4;
  const int g = lane >> 4;
  const int r16 = lane & 15;
  const __bf16* arow = OutAct + (size_t)(mt * 16 + r16) * DHTOT;

  v8f acc[4] = {};
  for (int kk = 0; kk < DHTOT; kk += 32) {
    const v16bf a = load_a_bf16(arow + kk, g);
#pragma unroll
    for (int j = 0; j < 4; ++j) {
      const int c = (nt0 + j) * 16 + r16;
      const v16bf b = load_b_bf16(proj_wb + (size_t)c * DHTOT + kk, g);
      acc[j] = WMMA_BF16(a, b, acc[j]);
    }
  }
#pragma unroll
  for (int j = 0; j < 4; ++j) {
    const int c = (nt0 + j) * 16 + r16;          // output channel (0..383)
    const float pb = proj_b[c];
#pragma unroll
    for (int v = 0; v < 8; ++v) {
      const size_t row = (size_t)(mt * 16 + v + g * 8);
      out[row * CDIM + c] = acc[j][v] + pb;
    }
  }
}

// ---- host-side launch ------------------------------------------------------
extern "C" void kernel_launch(void* const* d_in, const int* in_sizes, int n_in,
                              void* d_out, int out_size, void* d_ws, size_t ws_size,
                              hipStream_t stream) {
  const float* x           = (const float*)d_in[0];
  const float* qkv_w       = (const float*)d_in[1];
  const float* qkv_b       = (const float*)d_in[2];
  const float* proj_w      = (const float*)d_in[3];
  const float* proj_b      = (const float*)d_in[4];
  const float* attn_biases = (const float*)d_in[5];
  const int*   bias_idxs   = (const int*)d_in[6];
  (void)in_sizes; (void)n_in; (void)out_size; (void)ws_size;

  // ws layout (bf16): Q 20.4MB | K 20.4MB | Vt 88.1MB | OutAct 77.1MB | weights 1.5MB
  char* ws = (char*)d_ws;
  const size_t qbytes = (size_t)BATCH * NHEAD * NPAD * KEYD * 2;   // 20,447,232
  const size_t vbytes = (size_t)BATCH * NHEAD * DHEAD * KPAD * 2;  // 88,080,384
  const size_t obytes = (size_t)ROWS * DHTOT * 2;                  // 77,070,336
  __bf16* Q       = (__bf16*)ws;
  __bf16* Kb      = (__bf16*)(ws + qbytes);
  __bf16* Vt      = (__bf16*)(ws + 2 * qbytes);
  __bf16* OutAct  = (__bf16*)(ws + 2 * qbytes + vbytes);
  __bf16* qkv_wb  = (__bf16*)(ws + 2 * qbytes + vbytes + obytes);
  __bf16* proj_wb = qkv_wb + (size_t)HTOT * CDIM;

  cvt_kernel<<<(HTOT * CDIM) / 256, 256, 0, stream>>>(qkv_w, qkv_wb, HTOT * CDIM);
  cvt_kernel<<<(CDIM * DHTOT) / 256, 256, 0, stream>>>(proj_w, proj_wb, CDIM * DHTOT);
  qkv_kernel<<<(MTILES * 18) / 8, 256, 0, stream>>>(x, qkv_wb, qkv_b, Q, Kb, Vt);
  attn_kernel<<<BATCH * NHEAD, 256, SMEM_TOTAL, stream>>>(
      Q, Kb, Vt, attn_biases, bias_idxs, OutAct);
  proj_kernel<<<(MTILES * 6) / 8, 256, 0, stream>>>(OutAct, proj_wb, proj_b, (float*)d_out);
}
